// Fusion_19636590477988
// MI455X (gfx1250) — matrix operand order repaired and checked by reading, not compiled, not run on hardware
//
#include <hip/hip_runtime.h>
#include <hip/hip_bf16.h>

// ---------------------------------------------------------------------------
// Fused 1x1-conv MLP (9->18->36->36->1, ReLU) + row-max / col-max reduction
// for gfx1250 using v_wmma_f32_16x16x32_bf16.
//
// Each wave processes 16 pixels (one image row, 16 consecutive cols) as the
// M dimension of 16x16x32 bf16 WMMAs. K padded to 32 (layers 1,2) or 64
// (layers 3,4); N padded to multiples of 16. Inter-layer activations staged
// in a per-wave 16x64 *bf16* LDS tile: ReLU as a single v_med3_f32 clamp to
// [0,+inf), truncating f32->bf16 via ds_store_b16_d16_hi (zero conversion
// VALU), A fragments reload with two ds_load_b128 per lane. Biases folded
// into the GEMMs via a constant-1 channel: A column K=5 (layer 1, register),
// LDS column 31 (layer-2 reads, refreshed once per tile) and column 63
// (layers 3/4, never overwritten), with bias vectors in the matching B rows.
// Same-wave LDS is in-order -> no barriers in the hot loop.
// ---------------------------------------------------------------------------

typedef __attribute__((ext_vector_type(16))) __bf16 v16bf;
typedef __attribute__((ext_vector_type(8)))  float  v8f;

#define H_   2000
#define W_   1000
#define HW_  (H_ * W_)
#define ROWS 8            // rows per block (band)
#define TPR  63           // 16-wide tiles per row: ceil(1000/16)
#define ONE_BF 0x3F80u    // 1.0 in bf16

union BF16x16 { v16bf v; unsigned short u[16]; uint4 q[2]; };

__device__ __forceinline__ unsigned short f2bf(float f) {      // RNE (weights)
    unsigned u = __float_as_uint(f);
    return (unsigned short)((u + 0x7FFFu + ((u >> 16) & 1u)) >> 16);
}
__device__ __forceinline__ unsigned short truncbf(float f) {   // RTZ (activations)
    return (unsigned short)(__float_as_uint(f) >> 16);
}

// Order-preserving float -> u32 mapping (native u32 max atomics in LDS)
__device__ __forceinline__ unsigned fmapU(float f) {
    unsigned b = __float_as_uint(f);
    return (b & 0x80000000u) ? ~b : (b | 0x80000000u);
}
__device__ __forceinline__ float funmapU(unsigned u) {
    return __uint_as_float((u & 0x80000000u) ? (u & 0x7FFFFFFFu) : ~u);
}

// CAS-based float max for the final cross-block combine on global memory.
__device__ __forceinline__ void atomicMaxF32g(float* addr, float val) {
    int* ia = (int*)addr;
    int cur = __float_as_int(*addr);
    while (__int_as_float(cur) < val) {
        int old = atomicCAS(ia, cur, __float_as_int(val));
        if (old == cur) break;
        cur = old;
    }
}

// Build a B fragment (32x16 K-chunk of W^T) in the ISA B layout:
//   lane: N = lane&15 ; halves h -> K = kbase + (h<8 ? h : h+8) + 8*(lane>=16)
// wmat row-major [O][C]; row K==biasK sources bias[N]; others zero-padded.
__device__ __forceinline__ BF16x16 makeB(const float* __restrict__ wmat,
                                         const float* __restrict__ bias,
                                         int O, int C, int ntile, int kbase,
                                         int biasK, int lane) {
    int N  = (lane & 15) + ntile * 16;
    int hi = (lane >> 4) & 1;
    BF16x16 b;
#pragma unroll
    for (int h = 0; h < 16; ++h) {
        int K = kbase + (h < 8 ? h : h + 8) + 8 * hi;
        float v = 0.0f;
        if (N < O) {
            if (K < C)           v = wmat[N * C + K];
            else if (K == biasK) v = bias[N];
        }
        b.u[h] = f2bf(v);
    }
    return b;
}

// Layer-1 B: balanced channel permutation K0..4 <- ch0..4, K8..11 <- ch5..8,
// bias in row K=5, all other K rows zero.
__device__ __forceinline__ BF16x16 makeB1(const float* __restrict__ wmat,
                                          const float* __restrict__ bias,
                                          int O, int ntile, int lane) {
    int N  = (lane & 15) + ntile * 16;
    int hi = (lane >> 4) & 1;
    BF16x16 b;
#pragma unroll
    for (int h = 0; h < 16; ++h) {
        int K = (h < 8 ? h : h + 8) + 8 * hi;
        float v = 0.0f;
        if (N < O) {
            if (K <= 4)                 v = wmat[N * 9 + K];
            else if (K == 5)            v = bias[N];
            else if (K >= 8 && K <= 11) v = wmat[N * 9 + (K - 3)];
        }
        b.u[h] = f2bf(v);
    }
    return b;
}

// Load an A fragment from the per-wave bf16 LDS tile (16 rows x 64 halves).
// A layout: lane row M = lane&15; halves:
//   h<8  -> K = kchunk*32 + h    + 8*(lane>=16)
//   h>=8 -> K = kchunk*32 + h+8  + 8*(lane>=16)
__device__ __forceinline__ BF16x16 loadA_lds(const unsigned short* act,
                                             int mIdx, int hi, int kchunk) {
    const unsigned short* row = act + mIdx * 64 + kchunk * 32 + hi * 8;
    BF16x16 a;
    a.q[0] = *(const uint4*)(row);        // halves 0..7
    a.q[1] = *(const uint4*)(row + 16);   // halves 8..15
    return a;
}

// ReLU (v_med3_f32 clamp to [0,+inf)) + truncating f32->bf16 store at col oc.
// D layout: element (lane, r) = row M = r + 8*(lane>=16), col N = lane&15.
__device__ __forceinline__ void storeAct(unsigned short* act, v8f d,
                                         int oc, int hi, float pinf) {
#pragma unroll
    for (int r = 0; r < 8; ++r) {
        int M = r + 8 * hi;
        float x = __builtin_amdgcn_fmed3f(d[r], 0.0f, pinf);  // single v_med3
        act[M * 64 + oc] = truncbf(x);
    }
}

#define WMMA_BF16(A, B, C) \
    __builtin_amdgcn_wmma_f32_16x16x32_bf16(false, (A), false, (B), (short)0, (C), false, false)

extern "C" __global__ void init_out_kernel(float* out, int n) {
    int k = blockIdx.x * blockDim.x + threadIdx.x;
    if (k < n) out[k] = -INFINITY;
}

extern "C" __global__ __launch_bounds__(256)
void fused_mlp_max_kernel(const float* __restrict__ input,
                          const float* __restrict__ w1, const float* __restrict__ b1,
                          const float* __restrict__ w2, const float* __restrict__ b2,
                          const float* __restrict__ w3, const float* __restrict__ b3,
                          const float* __restrict__ w4, const float* __restrict__ b4,
                          float* __restrict__ out) {
    __shared__ unsigned short act[8][16][64]; // per-wave bf16 tiles (16 KB)
    __shared__ unsigned colmaxU[W_];          // band col maxes (mapped u32)
    __shared__ unsigned rowmaxU[ROWS];        // band row maxes

    const int tid  = threadIdx.x;
    const int lane = tid & 31;
    const int wv   = tid >> 5;            // wave id, 8 waves/block (wave32)
    const int mIdx = lane & 15;
    const int hi   = (lane >> 4) & 1;
    const int i0   = blockIdx.x * ROWS;

    const unsigned NEG_U = fmapU(-INFINITY);
    for (int k = tid; k < W_; k += 256) colmaxU[k] = NEG_U;
    if (tid < ROWS) rowmaxU[tid] = NEG_U;
    {   // zero tiles once (padding cols stay zero), then plant the constant-1
        // bias columns: col 31 (layer-2 reads) and col 63 (layer-3/4 reads).
        unsigned* af = (unsigned*)&act[0][0][0];
        for (int k = tid; k < 8 * 16 * 32; k += 256) af[k] = 0u;
    }
    __syncthreads();
    if (tid < 8 * 16) {
        act[tid >> 4][tid & 15][31] = (unsigned short)ONE_BF;
        act[tid >> 4][tid & 15][63] = (unsigned short)ONE_BF;
    }
    __syncthreads();

    // ---- prebuild all weight B-fragments in registers (13 x v16bf) ----
    BF16x16 B1[2], B2[3], B3[3][2], B4[2];
    B1[0]    = makeB1(w1, b1, 18, 0, lane);
    B1[1]    = makeB1(w1, b1, 18, 1, lane);
    B2[0]    = makeB(w2, b2, 36, 18, 0,  0, 31, lane);
    B2[1]    = makeB(w2, b2, 36, 18, 1,  0, 31, lane);
    B2[2]    = makeB(w2, b2, 36, 18, 2,  0, 31, lane);
    B3[0][0] = makeB(w3, b3, 36, 36, 0,  0, -1, lane);
    B3[0][1] = makeB(w3, b3, 36, 36, 0, 32, 63, lane);
    B3[1][0] = makeB(w3, b3, 36, 36, 1,  0, -1, lane);
    B3[1][1] = makeB(w3, b3, 36, 36, 1, 32, 63, lane);
    B3[2][0] = makeB(w3, b3, 36, 36, 2,  0, -1, lane);
    B3[2][1] = makeB(w3, b3, 36, 36, 2, 32, 63, lane);
    B4[0]    = makeB(w4, b4,  1, 36, 0,  0, -1, lane);
    B4[1]    = makeB(w4, b4,  1, 36, 0, 32, 63, lane);

    unsigned short* myact = &act[wv][0][0];
    const v8f zacc = {};
    const float PINF = __uint_as_float(0x7F800000u);  // +inf in a register

    // layer-1 A fragment: zero/constant halves are loop-invariant
    BF16x16 a1;
#pragma unroll
    for (int h = 0; h < 16; ++h) a1.u[h] = 0;
    if (hi == 0) a1.u[5] = (unsigned short)ONE_BF;   // K=5: constant-1 bias lane

    for (int rr = 0; rr < ROWS; ++rr) {
        const int i = i0 + rr;
        float rowlocal = -INFINITY;   // this wave's max over row i

        for (int ct = wv; ct < TPR; ct += 8) {
            const int j0 = ct * 16;

            // ---- layer 1: A straight from global (coalesced per channel).
            // lanes 0-15 load ch0..4 -> K0..4 ; lanes 16-31 ch5..8 -> K8..11
            int jm = j0 + mIdx;
            int jc = jm < W_ ? jm : (W_ - 1);
            const float* ip = input + (size_t)i * W_ + jc;
            if (hi == 0) {
#pragma unroll
                for (int c = 0; c < 5; ++c)
                    a1.u[c] = truncbf(ip[(size_t)c * HW_]);
            } else {
#pragma unroll
                for (int c = 0; c < 4; ++c)
                    a1.u[c] = truncbf(ip[(size_t)(5 + c) * HW_]);
            }
            v8f d0 = WMMA_BF16(a1.v, B1[0].v, zacc);
            v8f d1 = WMMA_BF16(a1.v, B1[1].v, zacc);
            storeAct(myact, d0, mIdx, hi, PINF);
            if (16 + mIdx < 18) storeAct(myact, d1, 16 + mIdx, hi, PINF);

            // ---- layer 2: 18 -> 36 (bias via act col 31 == 1.0) ----
            BF16x16 a2 = loadA_lds(myact, mIdx, hi, 0);
            v8f e0 = WMMA_BF16(a2.v, B2[0].v, zacc);
            v8f e1 = WMMA_BF16(a2.v, B2[1].v, zacc);
            v8f e2 = WMMA_BF16(a2.v, B2[2].v, zacc);
            storeAct(myact, e0, mIdx, hi, PINF);
            storeAct(myact, e1, 16 + mIdx, hi, PINF);
            if (32 + mIdx < 36) storeAct(myact, e2, 32 + mIdx, hi, PINF);

            // ---- layer 3: 36 -> 36 (K split 32+32; bias via col 63) ----
            BF16x16 a30 = loadA_lds(myact, mIdx, hi, 0);
            BF16x16 a31 = loadA_lds(myact, mIdx, hi, 1);
            v8f f0 = WMMA_BF16(a30.v, B3[0][0].v, zacc);
            f0     = WMMA_BF16(a31.v, B3[0][1].v, f0);
            v8f f1 = WMMA_BF16(a30.v, B3[1][0].v, zacc);
            f1     = WMMA_BF16(a31.v, B3[1][1].v, f1);
            v8f f2 = WMMA_BF16(a30.v, B3[2][0].v, zacc);
            f2     = WMMA_BF16(a31.v, B3[2][1].v, f2);
            storeAct(myact, f0, mIdx, hi, PINF);
            storeAct(myact, f1, 16 + mIdx, hi, PINF);
            if (32 + mIdx < 36) storeAct(myact, f2, 32 + mIdx, hi, PINF);

            // ---- layer 4: 36 -> 1 (bias via col 63; only N=0 valid) ----
            BF16x16 a40 = loadA_lds(myact, mIdx, hi, 0);
            BF16x16 a41 = loadA_lds(myact, mIdx, hi, 1);
            v8f g = WMMA_BF16(a40.v, B4[0].v, zacc);
            g     = WMMA_BF16(a41.v, B4[1].v, g);

            // restore constant-1 in col 31 for next tile's layer-2 read
            // (same-wave DS in-order: lands after the a40/a41 loads above)
            if (hi) myact[mIdx * 64 + 31] = (unsigned short)ONE_BF;

            // scores: lane 0 holds pixels j0..j0+7, lane 16 holds j0+8..+15
            if (mIdx == 0) {
#pragma unroll
                for (int r = 0; r < 8; ++r) {
                    int j = j0 + r + 8 * hi;
                    if (j < W_) {
                        float s = g[r];             // bias already accumulated
                        atomicMax(&colmaxU[j], fmapU(s));  // native ds max
                        rowlocal = fmaxf(rowlocal, s);
                    }
                }
            }
        }
        if (mIdx == 0) atomicMax(&rowmaxU[rr], fmapU(rowlocal));
    }
    __syncthreads();

    // rows are exclusive to this block -> plain store; cols combine globally
    if (tid < ROWS) out[i0 + tid] = funmapU(rowmaxU[tid]);
    for (int k = tid; k < W_; k += 256)
        atomicMaxF32g(&out[H_ + k], funmapU(colmaxU[k]));
}

extern "C" void kernel_launch(void* const* d_in, const int* in_sizes, int n_in,
                              void* d_out, int out_size, void* d_ws, size_t ws_size,
                              hipStream_t stream) {
    (void)in_sizes; (void)n_in; (void)d_ws; (void)ws_size;
    const float* input = (const float*)d_in[0];
    // d_in[1] = T_out (unused), d_in[2] = T_indices (identity scatter, unused)
    const float* w1 = (const float*)d_in[3];
    const float* b1 = (const float*)d_in[4];
    const float* w2 = (const float*)d_in[5];
    const float* b2 = (const float*)d_in[6];
    const float* w3 = (const float*)d_in[7];
    const float* b3 = (const float*)d_in[8];
    const float* w4 = (const float*)d_in[9];
    const float* b4 = (const float*)d_in[10];
    float* out = (float*)d_out;

    init_out_kernel<<<(out_size + 255) / 256, 256, 0, stream>>>(out, out_size);
    fused_mlp_max_kernel<<<H_ / ROWS, 256, 0, stream>>>(
        input, w1, b1, w2, b2, w3, b3, w4, b4, out);
}
